// LocalSimilarity_13348758356369
// MI455X (gfx1250) — matrix operand order repaired and checked by reading, not compiled, not run on hardware
//
#include <hip/hip_runtime.h>
#include <hip/hip_bf16.h>

typedef __bf16 bf16_t;
typedef __attribute__((ext_vector_type(16))) __bf16 v16bf;
typedef __attribute__((ext_vector_type(8)))  __bf16 v8bf;
typedef __attribute__((ext_vector_type(8)))  float  v8f;

#define Bq 8
#define Nq 32
#define Cq 768
#define Tq 256
#define Kq 5
#define SIM_THRq 0.05f
#define PATCH_THRq 2.0f

static __device__ __forceinline__ unsigned long long pack_sv(float v, unsigned idx) {
  // v >= SIM_THR (>0) when valid; nonneg float bits are monotone as u32.
  return ((unsigned long long)__float_as_uint(v) << 32) | (unsigned long long)idx;
}

// ---------------------------------------------------------------------------
// Kernel 1: per-(b,n) 256x256xK=768 GEMM via v_wmma_f32_16x16x32_bf16 on RAW
// bf16 features (normalization is bilinear -> applied in the epilogue as
// rnt[t]*rns[s], computed from sum-of-squares accumulated during staging).
// Fused mask+threshold and dual max/argmax (t->s, s->t) via ds_max_u64.
// Single pass over src_feats (the 201 MB input) -> HBM-roofline bound.
// ---------------------------------------------------------------------------
__global__ __launch_bounds__(1024) void sim_wmma_kernel(
    const float* __restrict__ tar_feat, const float* __restrict__ src_feats,
    const float* __restrict__ tar_mask, const float* __restrict__ src_masks,
    unsigned long long* __restrict__ pk_t2s,
    unsigned long long* __restrict__ pk_s2t) {
  __shared__ __align__(16) bf16_t lds_a[Tq * 32];   // [t][k]  16 KB
  __shared__ __align__(16) bf16_t lds_b[Tq * 32];   // [s][k]  16 KB
  __shared__ unsigned long long sh_t2s[Tq];
  __shared__ unsigned long long sh_s2t[Tq];
  __shared__ float sh_na[Tq];                       // sumsq -> 1/||tf_t||
  __shared__ float sh_nb[Tq];                       // sumsq -> 1/||sf_s||

  const int bn   = blockIdx.x;
  const int b    = bn >> 5;                 // N == 32
  const int tid  = threadIdx.x;
  const int w    = tid >> 5;                // wave id 0..31
  const int lane = tid & 31;
  const int h    = lane >> 4;               // half-wave
  const int l16  = lane & 15;
  const int tt   = w & 15;                  // this wave's t-tile
  const int sb   = (w >> 4) * 8;            // first of 8 s-tiles
  const int ts   = tid & 255;               // staging row (t / s)
  const int k0   = (tid >> 8) * 8;          // staging k-group (0,8,16,24)

  if (tid < Tq) {
    sh_t2s[tid] = 0ull; sh_s2t[tid] = 0ull;
    sh_na[tid] = 0.f;   sh_nb[tid] = 0.f;
  }

  v8f acc[8];
  #pragma unroll
  for (int j = 0; j < 8; ++j)
    #pragma unroll
    for (int r = 0; r < 8; ++r) acc[j][r] = 0.0f;

  // Each staging thread owns column ts with k in [k0, k0+8); loads are
  // coalesced across the wave (consecutive ts), LDS store is one b128.
  const float* tfp = tar_feat  + (size_t)b  * Cq * Tq + ts;
  const float* sfp = src_feats + (size_t)bn * Cq * Tq + ts;
  float ssqa = 0.f, ssqb = 0.f;

  for (int c0 = 0; c0 < Cq; c0 += 32) {
    __syncthreads();
    v8bf pa, pb;
    #pragma unroll
    for (int i = 0; i < 8; ++i) {
      float va = tfp[(size_t)(c0 + k0 + i) * Tq];
      float vb = sfp[(size_t)(c0 + k0 + i) * Tq];
      ssqa += va * va;
      ssqb += vb * vb;
      pa[i] = (bf16_t)va;
      pb[i] = (bf16_t)vb;
    }
    if (c0 + 32 < Cq) {  // prefetch next chunk (global_prefetch_b8)
      __builtin_prefetch(tfp + (size_t)(c0 + 32 + k0) * Tq, 0, 0);
      __builtin_prefetch(sfp + (size_t)(c0 + 32 + k0) * Tq, 0, 0);
    }
    *(v8bf*)&lds_a[ts * 32 + k0] = pa;   // ds_store_b128
    *(v8bf*)&lds_b[ts * 32 + k0] = pb;   // ds_store_b128
    __syncthreads();

    // A fragment (16x32 bf16): lanes 0-15 hold K 0..7 / 16..23, lanes 16-31
    // hold K 8..15 / 24..31 -> two contiguous 8-element runs per lane.
    const bf16_t* abase = &lds_a[(tt * 16 + l16) * 32];
    v8bf alo = *(const v8bf*)(abase + 8 * h);
    v8bf ahi = *(const v8bf*)(abase + 16 + 8 * h);
    v16bf afrag;
    #pragma unroll
    for (int i = 0; i < 8; ++i) { afrag[i] = alo[i]; afrag[i + 8] = ahi[i]; }

    #pragma unroll
    for (int j = 0; j < 8; ++j) {
      // B fragment (32x16 bf16): lanes 0-15 hold K 0..7 / 16..23, lanes 16-31 K 8..15 / 24..31.
      const bf16_t* bbase = &lds_b[((sb + j) * 16 + l16) * 32 + 8 * h];
      v8bf blo = *(const v8bf*)(bbase);
      v8bf bhi = *(const v8bf*)(bbase + 16);
      v16bf bfrag;
      #pragma unroll
      for (int i = 0; i < 8; ++i) { bfrag[i] = blo[i]; bfrag[i + 8] = bhi[i]; }
      acc[j] = __builtin_amdgcn_wmma_f32_16x16x32_bf16(
          false, afrag, false, bfrag, (short)0, acc[j], false, false);
    }
  }

  // Reduce sum-of-squares partials (4 per row) and turn into inverse norms.
  atomicAdd(&sh_na[ts], ssqa);           // ds_add_f32
  atomicAdd(&sh_nb[ts], ssqb);
  __syncthreads();
  if (tid < Tq) {
    sh_na[tid] = 1.0f / fmaxf(sqrtf(sh_na[tid]), 1e-12f);
    sh_nb[tid] = 1.0f / fmaxf(sqrtf(sh_nb[tid]), 1e-12f);
  }
  __syncthreads();

  // Epilogue: fold inverse norms and masks into per-row / per-col factors.
  // C/D layout: VGPR r, lane -> (m = r + 8h, n = l16).
  float fa[8], fb[8];
  #pragma unroll
  for (int r = 0; r < 8; ++r) {
    int t = tt * 16 + r + 8 * h;
    fa[r] = sh_na[t] * tar_mask[b * Tq + t];
  }
  #pragma unroll
  for (int j = 0; j < 8; ++j) {
    int s = (sb + j) * 16 + l16;
    fb[j] = sh_nb[s] * src_masks[(size_t)bn * Tq + s];
  }
  #pragma unroll
  for (int j = 0; j < 8; ++j)
    #pragma unroll
    for (int r = 0; r < 8; ++r) {
      float v = acc[j][r] * fa[r] * fb[j];
      if (v < SIM_THRq) v = 0.0f;
      acc[j][r] = v;
    }

  // t->s: per t, max over this wave's 8 s-tiles in registers, then ds_max_u64.
  #pragma unroll
  for (int r = 0; r < 8; ++r) {
    int t = tt * 16 + r + 8 * h;
    unsigned long long best = 0ull;
    #pragma unroll
    for (int j = 0; j < 8; ++j) {
      float v = acc[j][r];
      unsigned long long p = (v > 0.f) ? pack_sv(v, (unsigned)((sb + j) * 16 + l16)) : 0ull;
      best = (p > best) ? p : best;
    }
    atomicMax(&sh_t2s[t], best);
  }
  // s->t: per s, max over this wave's 8 t rows in registers, then ds_max_u64.
  #pragma unroll
  for (int j = 0; j < 8; ++j) {
    int s = (sb + j) * 16 + l16;
    unsigned long long best = 0ull;
    #pragma unroll
    for (int r = 0; r < 8; ++r) {
      float v = acc[j][r];
      unsigned long long p = (v > 0.f) ? pack_sv(v, (unsigned)(tt * 16 + r + 8 * h)) : 0ull;
      best = (p > best) ? p : best;
    }
    atomicMax(&sh_s2t[s], best);
  }
  __syncthreads();
  if (tid < Tq) {
    pk_t2s[(size_t)bn * Tq + tid] = sh_t2s[tid];
    pk_s2t[(size_t)bn * Tq + tid] = sh_s2t[tid];
  }
}

// ---------------------------------------------------------------------------
// Kernel 2: cycle consistency, mask_all, match_counts, sim_avg per (b,n).
// ---------------------------------------------------------------------------
__global__ __launch_bounds__(256) void postproc_kernel(
    const unsigned long long* __restrict__ pk_t2s,
    const unsigned long long* __restrict__ pk_s2t,
    const float* __restrict__ src_masks, const float* __restrict__ tar_mask,
    float* __restrict__ mask_all, float* __restrict__ sim_avg,
    float* __restrict__ out_counts) {
  __shared__ float red_c[256];
  __shared__ float red_s[256];
  int bn = blockIdx.x, b = bn >> 5;
  int t = threadIdx.x;
  unsigned long long p = pk_t2s[(size_t)bn * Tq + t];
  float sc_t2s = __uint_as_float((unsigned)(p >> 32));
  int i_t2s = (int)(p & 0xffffffffu);
  unsigned long long q = pk_s2t[(size_t)bn * Tq + i_t2s];
  float sim_s2s = __uint_as_float((unsigned)(q >> 32));
  int i_s2s = (int)(q & 0xffffffffu);
  int i_s2t_t = (int)(pk_s2t[(size_t)bn * Tq + t] & 0xffffffffu);
  float dw = (float)(i_s2s & 15) - (float)(t & 15);
  float dh = (float)(i_s2s >> 4) - (float)(t >> 4);
  float dist = sqrtf(dw * dw + dh * dh);
  bool m_sim = sc_t2s >= SIM_THRq;
  bool m_cyc = (dist <= PATCH_THRq) && (sim_s2s >= SIM_THRq);
  float m_t2s = src_masks[(size_t)bn * Tq + i_t2s];
  float m_nz = tar_mask[b * Tq + t] * m_t2s *
               ((i_s2t_t != 0) ? 1.f : 0.f) * ((i_t2s != 0) ? 1.f : 0.f);
  float ma = (m_sim && m_cyc) ? m_nz : 0.f;
  mask_all[(size_t)bn * Tq + t] = ma;
  red_c[t] = ma;
  red_s[t] = sc_t2s * ma;
  __syncthreads();
  for (int off = 128; off > 0; off >>= 1) {
    if (t < off) { red_c[t] += red_c[t + off]; red_s[t] += red_s[t + off]; }
    __syncthreads();
  }
  if (t == 0) {
    float cnt = red_c[0];
    sim_avg[bn] = (cnt > 0.f) ? (red_s[0] / 256.0f) : 0.f;
    out_counts[bn] = cnt;             // match_counts, float output
  }
}

// ---------------------------------------------------------------------------
// Kernel 3: top-K template selection (first-max-wins, matching lax.top_k)
// and output formatting into the flat float output buffer.
// ---------------------------------------------------------------------------
__global__ __launch_bounds__(256) void topk_format_kernel(
    const unsigned long long* __restrict__ pk_t2s,
    const float* __restrict__ mask_all, const float* __restrict__ sim_avg,
    float* __restrict__ out) {
  __shared__ int ids[Kq];
  int b = blockIdx.x;
  if (threadIdx.x == 0) {
    unsigned taken = 0u;
    for (int k = 0; k < Kq; ++k) {
      float best = -3.0e38f; int bi = 0;
      for (int n = 0; n < Nq; ++n) {
        if (!(taken & (1u << n))) {
          float v = sim_avg[b * Nq + n];
          if (v > best) { best = v; bi = n; }
        }
      }
      taken |= 1u << bi;
      ids[k] = bi;
      out[b * Kq + k]      = (float)bi;   // id_src   @ 0
      out[40 + b * Kq + k] = best;        // score_src @ 40
    }
  }
  __syncthreads();
  int t = threadIdx.x;
  for (int k = 0; k < Kq; ++k) {
    int bn = b * Nq + ids[k];
    unsigned long long p = pk_t2s[(size_t)bn * Tq + t];
    float sc = __uint_as_float((unsigned)(p >> 32));
    int it = (int)(p & 0xffffffffu);
    float pm = mask_all[(size_t)bn * Tq + t];
    bool nz = pm != 0.f;
    int bk = b * Kq + k;
    out[80 + bk * Tq + t] = sc;                          // score_pts @ 80
    size_t ot = 10320 + ((size_t)bk * Tq + t) * 2;       // tar_pts
    out[ot]     = nz ? (float)(t & 15) : -1.f;
    out[ot + 1] = nz ? (float)(t >> 4) : -1.f;
    size_t os = 30800 + ((size_t)bk * Tq + t) * 2;       // src_pts
    out[os]     = nz ? (float)(it & 15) : -1.f;
    out[os + 1] = nz ? (float)(it >> 4) : -1.f;
  }
}

extern "C" void kernel_launch(void* const* d_in, const int* in_sizes, int n_in,
                              void* d_out, int out_size, void* d_ws, size_t ws_size,
                              hipStream_t stream) {
  const float* src_feats = (const float*)d_in[0];   // (B,N,C,P,P)
  const float* tar_feat  = (const float*)d_in[1];   // (B,C,P,P)
  const float* src_masks = (const float*)d_in[2];   // (B,N,P,P)
  const float* tar_mask  = (const float*)d_in[3];   // (B,P,P)
  float* out = (float*)d_out;
  char* ws = (char*)d_ws;
  // workspace carve-out (~1.26 MB total)
  unsigned long long* pk_t2s = (unsigned long long*)(ws);              // 512 KB
  unsigned long long* pk_s2t = (unsigned long long*)(ws + (512 << 10));// 512 KB
  float* mask_all = (float*)(ws + (1024 << 10));                       // 256 KB
  float* sim_avg  = (float*)(ws + (1280 << 10));                       //   1 KB

  sim_wmma_kernel<<<Bq * Nq, 1024, 0, stream>>>(tar_feat, src_feats,
                                                tar_mask, src_masks,
                                                pk_t2s, pk_s2t);
  postproc_kernel<<<Bq * Nq, 256, 0, stream>>>(pk_t2s, pk_s2t, src_masks, tar_mask,
                                               mask_all, sim_avg, out + 51280);
  topk_format_kernel<<<Bq, 256, 0, stream>>>(pk_t2s, mask_all, sim_avg, out);
}
